// TQLinear_70772471103930
// MI455X (gfx1250) — matrix operand order repaired and checked by reading, not compiled, not run on hardware
//
#include <hip/hip_runtime.h>

#define N_ROWS  8192
#define IN_DIM  4096
#define OUT_DIM 4096
#define BM 128
#define BN 128
#define BK 32
#define LDK 40            /* BK + 8 elements of padding against bank conflicts */
#define NKT (IN_DIM / BK) /* 128 K-stages */

typedef __attribute__((ext_vector_type(8)))  float        v8f;
typedef __attribute__((ext_vector_type(4)))  __bf16       bf16x4;
typedef __attribute__((ext_vector_type(8)))  __bf16       bf16x8;
typedef __attribute__((ext_vector_type(16))) __bf16       bf16x16;
typedef __attribute__((ext_vector_type(4)))  int          i32x4;
typedef __attribute__((ext_vector_type(8)))  int          i32x8;
typedef __attribute__((ext_vector_type(4)))  unsigned int u32x4;

#define GLOBAL_AS __attribute__((address_space(1)))
#define LDS_AS    __attribute__((address_space(3)))

#if __has_builtin(__builtin_amdgcn_tensor_load_to_lds)
#define HAVE_TDM 1
#else
#define HAVE_TDM 0
#endif
#if __has_builtin(__builtin_amdgcn_global_load_async_to_lds_b128)
#define HAVE_ASYNC_LDS 1
#else
#define HAVE_ASYNC_LDS 0
#endif

static __device__ __forceinline__ void stage_wait() {
#if HAVE_TDM
#if __has_builtin(__builtin_amdgcn_s_wait_tensorcnt)
  __builtin_amdgcn_s_wait_tensorcnt(0);
#else
  asm volatile("s_wait_tensorcnt 0x0" ::: "memory");
#endif
#elif HAVE_ASYNC_LDS
#if __has_builtin(__builtin_amdgcn_s_wait_asynccnt)
  __builtin_amdgcn_s_wait_asynccnt(0);
#else
  asm volatile("s_wait_asynccnt 0x0" ::: "memory");
#endif
#endif
}

// -------- Kernel 0: W_int (int32) -> bf16, exact since 0 <= w < 128
__global__ __launch_bounds__(256) void tq_wcvt_kernel(
    const int* __restrict__ Wq, __bf16* __restrict__ Wb)
{
  const size_t i = ((size_t)blockIdx.x * 256 + threadIdx.x) * 4;
  int4 w = *(const int4*)(Wq + i);
  bf16x4 o;
  o[0] = (__bf16)(float)w.x;
  o[1] = (__bf16)(float)w.y;
  o[2] = (__bf16)(float)w.z;
  o[3] = (__bf16)(float)w.w;
  *(bf16x4*)(Wb + i) = o;
}

// -------- Kernel 1: s[m] = sum_i x[m,i] * d_in[e,i]  (8 rows per 256-thread block)
__global__ __launch_bounds__(256) void tq_rowsum_kernel(
    const float* __restrict__ X, const float* __restrict__ din_all,
    const int* __restrict__ eidx, float* __restrict__ S)
{
  const int e = eidx[0];
  const float* din = din_all + (size_t)e * IN_DIM;
  const int row  = blockIdx.x * 8 + (threadIdx.x >> 5);
  const int lane = threadIdx.x & 31;
  const float* xr = X + (size_t)row * IN_DIM;
  float sum = 0.0f;
  for (int i = lane * 4; i < IN_DIM; i += 32 * 4) {
    float4 a = *(const float4*)(xr + i);
    float4 d = *(const float4*)(din + i);
    sum += a.x * d.x + a.y * d.y + a.z * d.z + a.w * d.w;
  }
  #pragma unroll
  for (int off = 16; off > 0; off >>= 1) sum += __shfl_xor(sum, off, 32);
  if (lane == 0) S[row] = sum;
}

// -------- Kernel 2: tiled bf16 split-precision WMMA GEMM + fused epilogue
__global__ __launch_bounds__(256) void tq_gemm_kernel(
    const float* __restrict__ X, const __bf16* __restrict__ Wb,
    const float* __restrict__ zp, const float* __restrict__ dout_all,
    const float* __restrict__ din_all, const float* __restrict__ bias,
    const int* __restrict__ eidx, const float* __restrict__ S,
    float* __restrict__ Y)
{
  __shared__ __bf16 Ahi[2][BM * LDK];   // bf16(xs) high part
  __shared__ __bf16 Alo[2][BM * LDK];   // bf16 residual
  __shared__ __bf16 Bs [2][BN * LDK];   // bf16(W_int) tile (TDM / async copy)

  const int e = eidx[0];
  const float* din  = din_all  + (size_t)e * IN_DIM;
  const float* dout = dout_all + (size_t)e * OUT_DIM;

  const int tid   = threadIdx.x;
  const int lane  = tid & 31;
  const int wv    = tid >> 5;
  const int waveM = wv & 3;     // 4 row groups of 32
  const int waveN = wv >> 2;    // 2 col groups of 64
  const int m0 = blockIdx.y * BM;
  const int n0 = blockIdx.x * BN;

  float4 xr[4]; float4 dr[4];

  auto fetchX = [&](int kt) {
    const int kb = kt * BK;
    #pragma unroll
    for (int q = 0; q < 4; ++q) {
      int idx = tid + q * 256;   // 0..1023 -> 128 rows x 8 float4
      int row = idx >> 3;
      int c4  = idx & 7;
      xr[q] = *(const float4*)(X   + (size_t)(m0 + row) * IN_DIM + kb + c4 * 4);
      dr[q] = *(const float4*)(din + kb + c4 * 4);
    }
  };

  auto stageA = [&](int buf) {
    #pragma unroll
    for (int q = 0; q < 4; ++q) {
      int idx = tid + q * 256;
      int row = idx >> 3;
      int c4  = idx & 7;
      float xs0 = xr[q].x * dr[q].x, xs1 = xr[q].y * dr[q].y;
      float xs2 = xr[q].z * dr[q].z, xs3 = xr[q].w * dr[q].w;
      bf16x4 hv, lv;
      hv[0] = (__bf16)xs0; hv[1] = (__bf16)xs1; hv[2] = (__bf16)xs2; hv[3] = (__bf16)xs3;
      lv[0] = (__bf16)(xs0 - (float)hv[0]);
      lv[1] = (__bf16)(xs1 - (float)hv[1]);
      lv[2] = (__bf16)(xs2 - (float)hv[2]);
      lv[3] = (__bf16)(xs3 - (float)hv[3]);
      *(bf16x4*)(&Ahi[buf][row * LDK + c4 * 4]) = hv;
      *(bf16x4*)(&Alo[buf][row * LDK + c4 * 4]) = lv;
    }
  };

  // B tile: 128 rows x 32 bf16 (64B/row), LDS rows padded to 80B.
  auto issueB = [&](int kt, int buf) {
#if HAVE_TDM
    if (tid < 32) {   // one wave issues one TDM descriptor; EXEC ignored by TDM
      const unsigned long long ga =
          (unsigned long long)(const void*)(Wb + (size_t)n0 * IN_DIM + (size_t)kt * BK);
      const unsigned lds_off =
          (unsigned)(unsigned long long)(LDS_AS __bf16*)&Bs[buf][0];
      u32x4 g0;
      g0[0] = 1u;                                            // count=1, is_restore=0, gather=0
      g0[1] = lds_off;                                       // lds_addr [63:32]
      g0[2] = (unsigned)(ga & 0xFFFFFFFFu);                  // global_addr lo
      g0[3] = (unsigned)((ga >> 32) & 0x01FFFFFFu)           // global_addr [56:32]
            | (2u << 30);                                    // type=2 ("image")
      i32x8 g1;
      g1[0] = (int)((1u << 16)      /* data_size = 2B        */
                  | (1u << 20)      /* pad_enable            */
                  | (3u << 22)      /* pad_interval: 16 DW   */
                  | (3u << 25));    /* pad_amount: 4 DW=16B  */
      g1[1] = (int)(((unsigned)IN_DIM  & 0xFFFFu) << 16);    // tensor_dim0[15:0] -> bits 63:48
      g1[2] = (int)(((unsigned)OUT_DIM & 0xFFFFu) << 16);    // dim0 hi=0 | tensor_dim1 lo -> 95:80
      g1[3] = (int)((unsigned)BK << 16);                     // dim1 hi=0 | tile_dim0 -> 127:112
      g1[4] = (int)BN;                                       // tile_dim1 | tile_dim2=0
      g1[5] = (int)IN_DIM;                                   // tensor_dim0_stride[31:0]
      g1[6] = 0;                                             // stride0 hi | stride1 lo
      g1[7] = 0;
      i32x4 gz4 = {0, 0, 0, 0};
      i32x8 gz8 = {0, 0, 0, 0, 0, 0, 0, 0};
      __builtin_amdgcn_tensor_load_to_lds(g0, g1, gz4, gz4, gz8, 0);
    }
#else
    const int kb = kt * BK;
    #pragma unroll
    for (int q = 0; q < 2; ++q) {
      int chunk = tid + q * 256;          // 0..511
      int row = chunk >> 2;
      int c   = chunk & 3;                // 16B (8 bf16) units
      const __bf16* g = Wb + (size_t)(n0 + row) * IN_DIM + kb + c * 8;
      __bf16* l = &Bs[buf][row * LDK + c * 8];
#if HAVE_ASYNC_LDS
      __builtin_amdgcn_global_load_async_to_lds_b128(
          (GLOBAL_AS i32x4*)(unsigned long long)g,
          (LDS_AS i32x4*)l, 0, 0);
#else
      *(uint4*)l = *(const uint4*)g;
#endif
    }
#endif
  };

  v8f acc[2][4];
  const v8f vzero = {0.f,0.f,0.f,0.f,0.f,0.f,0.f,0.f};
  #pragma unroll
  for (int i = 0; i < 2; ++i)
    #pragma unroll
    for (int j = 0; j < 4; ++j) acc[i][j] = vzero;

  issueB(0, 0);
  fetchX(0);
  stageA(0);
  stage_wait();
  __syncthreads();

  const int l15   = lane & 15;
  const int khalf = (lane >> 4) * 8;    // A: half-wave K split (0..7/8..15, +16 in VGPR4-7)
  const int kb16  = (lane >> 4) * 16;   // unused: B also reads with khalf, same K split as A

  for (int kt = 0; kt < NKT; ++kt) {
    const int cur = kt & 1;
    if (kt + 1 < NKT) {
      issueB(kt + 1, cur ^ 1);
      fetchX(kt + 1);
    }
    if (kt + 2 < NKT) {
      __builtin_prefetch(X  + (size_t)(m0 + (tid >> 3)) * IN_DIM + (kt + 2) * BK + (tid & 7) * 4, 0, 1);
      __builtin_prefetch(Wb + (size_t)(n0 + (tid >> 2)) * IN_DIM + (kt + 2) * BK + (tid & 3) * 8, 0, 1);
    }

    bf16x16 fa_hi[2], fa_lo[2];
    #pragma unroll
    for (int rt = 0; rt < 2; ++rt) {
      const int row = waveM * 32 + rt * 16 + l15;
      const __bf16* pa = &Ahi[cur][row * LDK + khalf];
      bf16x8 a0 = *(const bf16x8*)pa;
      bf16x8 a1 = *(const bf16x8*)(pa + 16);
      fa_hi[rt] = __builtin_shufflevector(a0, a1, 0,1,2,3,4,5,6,7,8,9,10,11,12,13,14,15);
      const __bf16* pb = &Alo[cur][row * LDK + khalf];
      bf16x8 b0 = *(const bf16x8*)pb;
      bf16x8 b1 = *(const bf16x8*)(pb + 16);
      fa_lo[rt] = __builtin_shufflevector(b0, b1, 0,1,2,3,4,5,6,7,8,9,10,11,12,13,14,15);
    }

    #pragma unroll
    for (int ct = 0; ct < 4; ++ct) {
      const int wrow = waveN * 64 + ct * 16 + l15;
      const __bf16* pw = &Bs[cur][wrow * LDK + khalf];
      bf16x8 c0 = *(const bf16x8*)pw;
      bf16x8 c1 = *(const bf16x8*)(pw + 16);
      bf16x16 fb = __builtin_shufflevector(c0, c1, 0,1,2,3,4,5,6,7,8,9,10,11,12,13,14,15);
      #pragma unroll
      for (int rt = 0; rt < 2; ++rt) {
        acc[rt][ct] = __builtin_amdgcn_wmma_f32_16x16x32_bf16(
            false, fa_hi[rt], false, fb, (short)0, acc[rt][ct], false, false);
        acc[rt][ct] = __builtin_amdgcn_wmma_f32_16x16x32_bf16(
            false, fa_lo[rt], false, fb, (short)0, acc[rt][ct], false, false);
      }
    }

    __syncthreads();
    if (kt + 1 < NKT) stageA(cur ^ 1);
    stage_wait();
    __syncthreads();
  }

  // Epilogue: y[m,o] = dout[o] * (acc - zp[o]*s[m]) + bias[o]
  #pragma unroll
  for (int rt = 0; rt < 2; ++rt) {
    const int mBase = m0 + waveM * 32 + rt * 16 + ((lane >> 4) << 3);
    float sv[8];
    #pragma unroll
    for (int r = 0; r < 8; ++r) sv[r] = S[mBase + r];
    #pragma unroll
    for (int ct = 0; ct < 4; ++ct) {
      const int n = n0 + waveN * 64 + ct * 16 + l15;
      const float sc = dout[n], zv = zp[n], bz = bias[n];
      #pragma unroll
      for (int r = 0; r < 8; ++r) {
        Y[(size_t)(mBase + r) * OUT_DIM + n] = sc * (acc[rt][ct][r] - zv * sv[r]) + bz;
      }
    }
  }
}

extern "C" void kernel_launch(void* const* d_in, const int* in_sizes, int n_in,
                              void* d_out, int out_size, void* d_ws, size_t ws_size,
                              hipStream_t stream) {
  const float* x      = (const float*)d_in[0];
  const int*   wint   = (const int*)d_in[1];
  const float* zp     = (const float*)d_in[2];
  const float* dout_e = (const float*)d_in[3];
  const float* din_e  = (const float*)d_in[4];
  const float* bias   = (const float*)d_in[5];
  const int*   eidx   = (const int*)d_in[6];
  float* y = (float*)d_out;

  float*  S  = (float*)d_ws;                        // 32 KB
  __bf16* Wb = (__bf16*)((char*)d_ws + 32768);      // 32 MB pre-converted weights

  tq_wcvt_kernel<<<(OUT_DIM * (size_t)IN_DIM) / (256 * 4), 256, 0, stream>>>(wint, Wb);
  tq_rowsum_kernel<<<N_ROWS / 8, 256, 0, stream>>>(x, din_e, eidx, S);
  dim3 grid(OUT_DIM / BN, N_ROWS / BM);   // (32, 64)
  tq_gemm_kernel<<<grid, 256, 0, stream>>>(x, Wb, zp, dout_e, din_e, bias, eidx, S, y);
}
